// CCR_59193239273568
// MI455X (gfx1250) — matrix operand order repaired and checked by reading, not compiled, not run on hardware
//
#include <hip/hip_runtime.h>
#include <math.h>

typedef __attribute__((ext_vector_type(16))) _Float16 v16h;
typedef __attribute__((ext_vector_type(8)))  float    v8f;

#define BB  2
#define CIN 64
#define CC  32
#define HH  64
#define WW  64
#define NN  (HH * WW)   // 4096

// ---------------------------------------------------------------------------
// DPP 16-lane butterfly reductions (pure VALU, no LDS):
//   0xB1  = quad_perm(1,0,3,2)  -> xor 1
//   0x4E  = quad_perm(2,3,0,1)  -> xor 2
//   0x141 = row_half_mirror     -> xor 7 (== xor 4 on quad-uniform values)
//   0x140 = row_mirror          -> xor 15 (== xor 8 on 8-group-uniform values)
// ---------------------------------------------------------------------------
template <int CTRL>
__device__ __forceinline__ float dpp_mov(float x) {
  return __builtin_bit_cast(float,
      __builtin_amdgcn_update_dpp(0, __builtin_bit_cast(int, x), CTRL, 0xF, 0xF, true));
}
__device__ __forceinline__ float red_max16(float x) {
  x = fmaxf(x, dpp_mov<0xB1>(x));
  x = fmaxf(x, dpp_mov<0x4E>(x));
  x = fmaxf(x, dpp_mov<0x141>(x));
  x = fmaxf(x, dpp_mov<0x140>(x));
  return x;
}
__device__ __forceinline__ float red_sum16(float x) {
  x += dpp_mov<0xB1>(x);
  x += dpp_mov<0x4E>(x);
  x += dpp_mov<0x141>(x);
  x += dpp_mov<0x140>(x);
  return x;
}

// ---------------------------------------------------------------------------
// Direct 3x3 conv (SAME, stride 1) + bias + ReLU.  dst is [B][32][H][W].
// ~2.4 GFLOP total across all calls -> VALU is fine; WMMA is spent on the
// attention GEMMs where the FLOPs live.
// ---------------------------------------------------------------------------
__global__ __launch_bounds__(256)
void conv3_relu_kernel(const float* __restrict__ src, const float* __restrict__ wgt,
                       const float* __restrict__ bias, float* __restrict__ dst, int Cin)
{
  int idx = blockIdx.x * blockDim.x + threadIdx.x;   // [B][32][H][W] flat
  int xw = idx & 63;
  int yh = (idx >> 6) & 63;
  int co = (idx >> 12) & 31;
  int b  = idx >> 17;
  float acc = bias[co];
  const float* wp = wgt + (size_t)co * Cin * 9;
  for (int ci = 0; ci < Cin; ++ci) {
    const float* sp = src + ((size_t)(b * Cin + ci) * HH) * WW;
    const float* wq = wp + ci * 9;
    #pragma unroll
    for (int dy = -1; dy <= 1; ++dy) {
      int y = yh + dy;
      if (y < 0 || y >= HH) continue;
      #pragma unroll
      for (int dx = -1; dx <= 1; ++dx) {
        int x = xw + dx;
        if (x < 0 || x >= WW) continue;
        acc += sp[y * WW + x] * wq[(dy + 1) * 3 + (dx + 1)];
      }
    }
  }
  dst[idx] = fmaxf(acc, 0.0f);
}

// ---------------------------------------------------------------------------
// Repack f32 [B][C][N] into three fragment-ready f16 layouts so the flash
// kernel issues only contiguous 32-byte per-lane loads (global_load_b128 x2):
//   Apk: WMMA A-operand tiles (16 query rows x K=32 channels), scale folded in
//   Kpk: WMMA B-operand tiles for S = Q·K      (K rows = channels)
//   Vpk: WMMA B-operand tiles for ctx = P·V    (K rows = key index)
// A-layout lane/half mapping (ISA 16-bit A 16x32):
//   g(c) = (c>>3)&1,  j(c) = (c&7) + ((c>>4)<<3)
// ---------------------------------------------------------------------------
__global__ __launch_bounds__(256)
void pack_frag_kernel(const float* __restrict__ src, _Float16* __restrict__ Apk,
                      _Float16* __restrict__ Kpk, _Float16* __restrict__ Vpk,
                      float ascale)
{
  int idx = blockIdx.x * blockDim.x + threadIdx.x;   // [B][C][N] flat
  int n = idx & (NN - 1);
  int c = (idx >> 12) & 31;
  int b = idx >> 17;
  float v = src[idx];
  _Float16 h  = (_Float16)v;
  _Float16 ha = (_Float16)(v * ascale);

  // A fragment: tile = n>>4, lane = (n&15) | g(c)<<4, half = j(c)
  int gA = (c >> 3) & 1;
  int jA = (c & 7) + ((c >> 4) << 3);
  int laneA = (n & 15) | (gA << 4);
  Apk[(((size_t)b * 256 + (n >> 4)) * 32 + laneA) * 16 + jA] = ha;

  // K B-operand: tile = n>>4, lane = (n&15) | g(c)<<4, half = j(c)
  int laneK = (n & 15) | (gA << 4);
  Kpk[(((size_t)b * 256 + (n >> 4)) * 32 + laneK) * 16 + jA] = h;

  // V B-operand: chunk = n>>5, col-half = c>>4,
  //              lane = (c&15) | ((n>>3)&1)<<4, half = (n&7) + ((n>>4)&1)*8
  int laneV = (c & 15) | (((n >> 3) & 1) << 4);
  Vpk[((((size_t)b * 128 + (n >> 5)) * 2 + (c >> 4)) * 32 + laneV) * 16 + ((n & 7) + (((n >> 4) & 1) << 3))] = h;
}

// ---------------------------------------------------------------------------
// Fused flash attention pass:
//   ctx[b,c,n] = softmax_m(scale * A[n,:]·B[:,m]) · V[m,c]
// One wave32 owns 16 query rows.  64 keys per iteration:
//   4x  v_wmma_f32_16x16x32_f16  (S tiles, K = C = 32)
//   4x  v_wmma_f32_16x16x32_f16  (ctx += P·V, K = 32)
// Online softmax (DPP reductions) keeps the N x N similarity matrix on-chip
// (saves ~1.6 GB of HBM traffic vs materializing it at 23.3 TB/s).
// P is staged through LDS directly into A-fragment positions so the re-load
// is two contiguous ds_load pairs per lane.
// ---------------------------------------------------------------------------
__global__ __launch_bounds__(256)
void flash_attn_kernel(const _Float16* __restrict__ Apk,
                       const _Float16* __restrict__ Kpk,
                       const _Float16* __restrict__ Vpk,
                       float* __restrict__ ctx)
{
  __shared__ _Float16 pshm[8][32][40];     // 40-half (80B) row stride: 16B-aligned
  int wslot = threadIdx.x >> 5;
  int lane  = threadIdx.x & 31;
  int wave  = (blockIdx.x << 3) + wslot;   // 512 waves total
  int b     = wave >> 8;
  int tile  = wave & 255;                  // 16-row query tile
  int col   = lane & 15;
  int grp   = lane >> 4;

  const _Float16* Kb = Kpk + (size_t)b * 131072;
  const _Float16* Vb = Vpk + (size_t)b * 131072;

  // A fragment (scale pre-folded): one contiguous 32B load per lane
  v16h a = *(const v16h*)(Apk + (((size_t)b * 256 + tile) * 32 + lane) * 16);

  v8f acc0 = {};
  v8f acc1 = {};
  float rmax[8], rsum[8];
  #pragma unroll
  for (int r = 0; r < 8; ++r) { rmax[r] = -3.0e38f; rsum[r] = 0.0f; }

  // P staging targets: all four S tiles of a row land in the same LDS row
  int gprime = (col >> 3) & 1;
  int jlo    = col & 7;

  for (int m = 0; m < NN; m += 64) {
    // ---- S tiles: 4 key tiles of 16 columns, fragment-ready loads --------
    int t0 = m >> 4;
    v16h kb0 = *(const v16h*)(Kb + (((size_t)(t0 + 0)) * 32 + lane) * 16);
    v16h kb1 = *(const v16h*)(Kb + (((size_t)(t0 + 1)) * 32 + lane) * 16);
    v16h kb2 = *(const v16h*)(Kb + (((size_t)(t0 + 2)) * 32 + lane) * 16);
    v16h kb3 = *(const v16h*)(Kb + (((size_t)(t0 + 3)) * 32 + lane) * 16);
    v8f z = {};
    v8f s0 = __builtin_amdgcn_wmma_f32_16x16x32_f16(false, a, false, kb0, (short)0, z, false, false);
    v8f s1 = __builtin_amdgcn_wmma_f32_16x16x32_f16(false, a, false, kb1, (short)0, z, false, false);
    v8f s2 = __builtin_amdgcn_wmma_f32_16x16x32_f16(false, a, false, kb2, (short)0, z, false, false);
    v8f s3 = __builtin_amdgcn_wmma_f32_16x16x32_f16(false, a, false, kb3, (short)0, z, false, false);

    // ---- online softmax (row stats via DPP, no LDS) ----------------------
    #pragma unroll
    for (int r = 0; r < 8; ++r) {
      float v0 = s0[r], v1 = s1[r], v2 = s2[r], v3 = s3[r];
      float cm = red_max16(fmaxf(fmaxf(v0, v1), fmaxf(v2, v3)));
      float nm = fmaxf(rmax[r], cm);
      float cr = __expf(rmax[r] - nm);          // first chunk: exp(-inf) = 0
      rmax[r] = nm;
      float p0 = __expf(v0 - nm);
      float p1 = __expf(v1 - nm);
      float p2 = __expf(v2 - nm);
      float p3 = __expf(v3 - nm);
      float ps = red_sum16((p0 + p1) + (p2 + p3));
      rsum[r] = rsum[r] * cr + ps;
      acc0[r] *= cr;
      acc1[r] *= cr;
      s0[r] = p0; s1[r] = p1; s2[r] = p2; s3[r] = p3;
    }

    // ---- store P directly into A-fragment positions in LDS ---------------
    #pragma unroll
    for (int r = 0; r < 8; ++r) {
      _Float16* pp = &pshm[wslot][(r + 8 * grp) + 16 * gprime][0];
      pp[jlo]      = (_Float16)s0[r];
      pp[jlo + 8]  = (_Float16)s1[r];
      pp[jlo + 16] = (_Float16)s2[r];
      pp[jlo + 24] = (_Float16)s3[r];
    }
    __syncthreads();                        // uniform trip count: legal
    v16h pa0 = *(const v16h*)&pshm[wslot][lane][0];
    v16h pa1 = *(const v16h*)&pshm[wslot][lane][16];
    __syncthreads();

    // ---- ctx += P(16x64) · V(64x32): four K=32 WMMAs ---------------------
    size_t mc = (size_t)(m >> 5);
    v16h vb00 = *(const v16h*)(Vb + (((mc + 0) * 2 + 0) * 32 + lane) * 16);
    v16h vb01 = *(const v16h*)(Vb + (((mc + 0) * 2 + 1) * 32 + lane) * 16);
    v16h vb10 = *(const v16h*)(Vb + (((mc + 1) * 2 + 0) * 32 + lane) * 16);
    v16h vb11 = *(const v16h*)(Vb + (((mc + 1) * 2 + 1) * 32 + lane) * 16);
    acc0 = __builtin_amdgcn_wmma_f32_16x16x32_f16(false, pa0, false, vb00, (short)0, acc0, false, false);
    acc0 = __builtin_amdgcn_wmma_f32_16x16x32_f16(false, pa1, false, vb10, (short)0, acc0, false, false);
    acc1 = __builtin_amdgcn_wmma_f32_16x16x32_f16(false, pa0, false, vb01, (short)0, acc1, false, false);
    acc1 = __builtin_amdgcn_wmma_f32_16x16x32_f16(false, pa1, false, vb11, (short)0, acc1, false, false);
  }

  // epilogue: divide by softmax denominator, write channel-major [B][C][N]
  float* cb0 = ctx + (size_t)b * CC * NN;
  int n0 = tile << 4;
  #pragma unroll
  for (int r = 0; r < 8; ++r) {
    int n = n0 + r + 8 * grp;
    float inv = 1.0f / rsum[r];
    cb0[(size_t)col * NN + n]        = acc0[r] * inv;
    cb0[(size_t)(16 + col) * NN + n] = acc1[r] * inv;
  }
}

// ---------------------------------------------------------------------------
// cat = concat([x, (c1+c2+c3)/3], channel)  ->  [B][96][N]
// ---------------------------------------------------------------------------
__global__ __launch_bounds__(256)
void cat_avg_kernel(const float* __restrict__ x, const float* __restrict__ c1,
                    const float* __restrict__ c2, const float* __restrict__ c3,
                    float* __restrict__ cat)
{
  int idx = blockIdx.x * blockDim.x + threadIdx.x;   // [B][96][N] flat
  int n  = idx & (NN - 1);
  int ch = (idx >> 12) % 96;
  int b  = idx / (96 * NN);
  if (ch < CIN) {
    cat[idx] = x[((size_t)b * CIN + ch) * NN + n];
  } else {
    size_t o = ((size_t)b * CC + (ch - CIN)) * NN + n;
    cat[idx] = (c1[o] + c2[o] + c3[o]) * (1.0f / 3.0f);
  }
}

// ---------------------------------------------------------------------------
extern "C" void kernel_launch(void* const* d_in, const int* in_sizes, int n_in,
                              void* d_out, int out_size, void* d_ws, size_t ws_size,
                              hipStream_t stream)
{
  (void)in_sizes; (void)n_in; (void)out_size; (void)ws_size;
  const float* x   = (const float*)d_in[0];
  const float* wq1 = (const float*)d_in[1];  const float* bq1 = (const float*)d_in[2];
  const float* wq2 = (const float*)d_in[3];  const float* bq2 = (const float*)d_in[4];
  const float* wk1 = (const float*)d_in[5];  const float* bk1 = (const float*)d_in[6];
  const float* wk2 = (const float*)d_in[7];  const float* bk2 = (const float*)d_in[8];
  const float* wv1 = (const float*)d_in[9];  const float* bv1 = (const float*)d_in[10];
  const float* wv2 = (const float*)d_in[11]; const float* bv2 = (const float*)d_in[12];
  const float* wr  = (const float*)d_in[13]; const float* br  = (const float*)d_in[14];
  const float* wg  = (const float*)d_in[15]; const float* bg  = (const float*)d_in[16];
  const float* wb  = (const float*)d_in[17]; const float* bb  = (const float*)d_in[18];
  const float* w2  = (const float*)d_in[19]; const float* b2  = (const float*)d_in[20];
  const float* w3  = (const float*)d_in[21]; const float* b3  = (const float*)d_in[22];

  char* ws = (char*)d_ws;
  const size_t S32 = (size_t)BB * CC * NN * sizeof(float);      // 1 MB
  const size_t S16 = (size_t)BB * CC * NN * sizeof(_Float16);   // 0.5 MB
  float*    t0   = (float*)(ws);
  float*    qf   = (float*)(ws + 1 * S32);
  float*    kf   = (float*)(ws + 2 * S32);
  float*    vf   = (float*)(ws + 3 * S32);
  _Float16* qA   = (_Float16*)(ws + 4 * S32);
  _Float16* qK   = (_Float16*)(ws + 4 * S32 + 1 * S16);
  _Float16* qV   = (_Float16*)(ws + 4 * S32 + 2 * S16);
  _Float16* kA   = (_Float16*)(ws + 4 * S32 + 3 * S16);
  _Float16* kK   = (_Float16*)(ws + 4 * S32 + 4 * S16);
  _Float16* kV   = (_Float16*)(ws + 4 * S32 + 5 * S16);
  _Float16* vA   = (_Float16*)(ws + 4 * S32 + 6 * S16);
  _Float16* vK   = (_Float16*)(ws + 4 * S32 + 7 * S16);
  _Float16* vV   = (_Float16*)(ws + 4 * S32 + 8 * S16);
  char*     p2   = ws + 4 * S32 + 9 * S16;
  float*    ctx1 = (float*)(p2);
  float*    ctx2 = (float*)(p2 + 1 * S32);
  float*    ctx3 = (float*)(p2 + 2 * S32);
  float*    ca   = (float*)(p2 + 3 * S32);
  float*    cb   = (float*)(p2 + 4 * S32);
  float*    cc   = (float*)(p2 + 5 * S32);
  float*    catb = (float*)(p2 + 6 * S32);                       // 3 MB
  float*    t1   = (float*)(p2 + 9 * S32);
  float*    out  = (float*)d_out;

  const int convGrid = (BB * CC * HH * WW) / 256;   // 1024
  const float scale = 0.17677669529663687f;         // 32^-0.5

  // Q / K / V towers: conv3+relu twice each
  conv3_relu_kernel<<<convGrid, 256, 0, stream>>>(x,  wq1, bq1, t0, CIN);
  conv3_relu_kernel<<<convGrid, 256, 0, stream>>>(t0, wq2, bq2, qf, CC);
  conv3_relu_kernel<<<convGrid, 256, 0, stream>>>(x,  wk1, bk1, t0, CIN);
  conv3_relu_kernel<<<convGrid, 256, 0, stream>>>(t0, wk2, bk2, kf, CC);
  conv3_relu_kernel<<<convGrid, 256, 0, stream>>>(x,  wv1, bv1, t0, CIN);
  conv3_relu_kernel<<<convGrid, 256, 0, stream>>>(t0, wv2, bv2, vf, CC);

  // fragment-ready f16 operand layouts (scale folded into the A copies)
  pack_frag_kernel<<<convGrid, 256, 0, stream>>>(qf, qA, qK, qV, scale);
  pack_frag_kernel<<<convGrid, 256, 0, stream>>>(kf, kA, kK, kV, scale);
  pack_frag_kernel<<<convGrid, 256, 0, stream>>>(vf, vA, vK, vV, scale);

  // three fused attention passes (512 waves = 64 blocks x 8 waves)
  flash_attn_kernel<<<64, 256, 0, stream>>>(qA, kK, vV, ctx1);  // softmax(qT k) vT
  flash_attn_kernel<<<64, 256, 0, stream>>>(kA, vK, qV, ctx2);  // softmax(kT v) qT
  flash_attn_kernel<<<64, 256, 0, stream>>>(vA, qK, kV, ctx3);  // softmax(vT q) kT

  // per-pass output convs
  conv3_relu_kernel<<<convGrid, 256, 0, stream>>>(ctx1, wr, br, ca, CC);
  conv3_relu_kernel<<<convGrid, 256, 0, stream>>>(ctx2, wg, bg, cb, CC);
  conv3_relu_kernel<<<convGrid, 256, 0, stream>>>(ctx3, wb, bb, cc, CC);

  // concat + average, then final two convs
  cat_avg_kernel<<<(BB * 96 * NN) / 256, 256, 0, stream>>>(x, ca, cb, cc, catb);
  conv3_relu_kernel<<<convGrid, 256, 0, stream>>>(catb, w2, b2, t1, CIN + CC);
  conv3_relu_kernel<<<convGrid, 256, 0, stream>>>(t1, w3, b3, out, CC);
}
